// DDGAttention_15547781611607
// MI455X (gfx1250) — matrix operand order repaired and checked by reading, not compiled, not run on hardware
//
#include <hip/hip_runtime.h>
#include <hip/hip_bf16.h>

typedef __attribute__((ext_vector_type(16))) _Float16 v16h;
typedef __attribute__((ext_vector_type(8)))  float    v8f;

#define NB    4
#define LSEQ  1024
#define DIN   128
#define DOUT  128
#define NH    12
#define DQK   16
#define DV    16
#define DFEAT 288   // 192 node + 84 spatial, padded to 9*32
#define INFV  100000.0f

__device__ __forceinline__ v8f wmma_f16(v16h a, v16h b, v8f c) {
  return __builtin_amdgcn_wmma_f32_16x16x32_f16(false, a, false, b, (short)0, c,
                                                false, false);
}

// ---------------------------------------------------------------------------
// Kernel A: QKV projection.  x(N,L,128) f32 -> q(N,L,H,16) f16,
// kT(N,H,16,L) f16 (transposed for contiguous K^T fragment loads),
// v(N,L,H,16) f16.  4 waves/block share one 16x128 x-tile via LDS.
// ---------------------------------------------------------------------------
__global__ __launch_bounds__(128) void qkv_proj_kernel(
    const float* __restrict__ x, const float* __restrict__ Wq,
    const float* __restrict__ Wk, const float* __restrict__ Wv,
    _Float16* __restrict__ qb, _Float16* __restrict__ kTb,
    _Float16* __restrict__ vb) {
  __shared__ _Float16 xs[16][DIN];
  int rt = blockIdx.x / 9;        // 256 row tiles (16 rows each)
  int cg = blockIdx.x % 9;        // group of 4 column tiles
  int n = rt >> 6;
  int sbase = (rt & 63) << 4;

  const float* xp = x + ((size_t)n * LSEQ + sbase) * DIN;
  for (int i = threadIdx.x; i < 16 * DIN; i += blockDim.x)
    xs[i / DIN][i % DIN] = (_Float16)xp[i];
  __syncthreads();

  int wave = threadIdx.x >> 5;
  int lane = threadIdx.x & 31;
  int ct = cg * 4 + wave;          // 0..35 : 12 tiles each for q,k,v
  int mat = ct / 12;               // 0=q 1=k 2=v
  int hct = ct % 12;               // head index (tile width == head width)
  const float* W = (mat == 0) ? Wq : (mat == 1) ? Wk : Wv;

  int hs = lane >> 4;              // half-wave select
  int m  = lane & 15;
  v8f acc{};
#pragma unroll
  for (int kk = 0; kk < 4; ++kk) {
    int kb = kk * 32;
    v16h a, b;
#pragma unroll
    for (int j = 0; j < 16; ++j) {
      int K = ((j < 8) ? j : (j + 8)) + hs * 8;
      a[j] = xs[m][kb + K];
    }
    const float* wp = W + (size_t)kb * (NH * DQK) + hct * 16 + m;
#pragma unroll
    for (int j = 0; j < 16; ++j) {
      int K = ((j < 8) ? j : (j + 8)) + hs * 8;
      b[j] = (_Float16)wp[(size_t)K * (NH * DQK)];
    }
    acc = wmma_f16(a, b, acc);
  }

  int d = lane & 15;
  int mo = hs * 8;
  if (mat == 1) {  // transposed K
    _Float16* kp = kTb + (((size_t)n * NH + hct) * DQK + d) * LSEQ + sbase;
#pragma unroll
    for (int r = 0; r < 8; ++r) kp[mo + r] = (_Float16)acc[r];
  } else {
    _Float16* ob = (mat == 0) ? qb : vb;
#pragma unroll
    for (int r = 0; r < 8; ++r)
      ob[(((size_t)n * LSEQ + sbase + mo + r) * NH + hct) * 16 + d] =
          (_Float16)acc[r];
  }
}

// ---------------------------------------------------------------------------
// Kernel B: flash attention per (n, head, 16-query tile), one wave/block.
// Streams 64 keys/iter: 4 WMMAs for logits, one online-softmax update
// (shuffle reductions amortized over 64 keys), P via LDS, then 4 WMMAs for
// the fused [V | pos_CB, 1] aggregation.  Never materializes L x L logits.
// ---------------------------------------------------------------------------
__global__ __launch_bounds__(32) void attn_kernel(
    const _Float16* __restrict__ qb, const _Float16* __restrict__ kTb,
    const _Float16* __restrict__ vb, const float* __restrict__ pos_CA,
    const float* __restrict__ pos_CB, const unsigned char* __restrict__ mask,
    _Float16* __restrict__ feat, float* __restrict__ apb) {
  __shared__ _Float16 Pl[16][64];

  int b = blockIdx.x;
  int qt = b & 63;
  int h  = (b >> 6) % NH;
  int n  = b / (64 * NH);
  int lane = threadIdx.x;
  int hs = lane >> 4, m = lane & 15, col = lane & 15, mo = hs * 8;
  int qbase = qt * 16;

  // Q fragment: A-layout 16x32, dqk padded 16->32 with zeros
  v16h aq;
  const _Float16* qp = qb + (((size_t)n * LSEQ + qbase + m) * NH + h) * 16;
#pragma unroll
  for (int j = 0; j < 16; ++j) {
    int K = ((j < 8) ? j : (j + 8)) + hs * 8;
    aq[j] = (K < DQK) ? qp[K] : (_Float16)0.f;
  }

  float mrow[8], lrow[8];
  v8f accV{}, accP{};
#pragma unroll
  for (int r = 0; r < 8; ++r) { mrow[r] = -1e30f; lrow[r] = 0.f; }

  const _Float16* kTh = kTb + ((size_t)n * NH + h) * DQK * LSEQ;
  const unsigned char* mkn = mask + (size_t)n * LSEQ;

  for (int kb = 0; kb < LSEQ; kb += 64) {
    if (kb + 64 < LSEQ && lane < DQK)
      __builtin_prefetch(kTh + (size_t)lane * LSEQ + kb + 64, 0, 1);

    // K^T B-fragments: 4 tiles of 16 keys (lane = key column, elements = dqk)
    v16h bk[4];
#pragma unroll
    for (int t = 0; t < 4; ++t)
#pragma unroll
      for (int j = 0; j < 16; ++j) {
        int K = ((j < 8) ? j : (j + 8)) + hs * 8;
        bk[t][j] = (K < DQK) ? kTh[(size_t)K * LSEQ + kb + t * 16 + col]
                             : (_Float16)0.f;
      }
    v8f z{};
    v8f s0 = wmma_f16(aq, bk[0], z);
    v8f s1 = wmma_f16(aq, bk[1], z);
    v8f s2 = wmma_f16(aq, bk[2], z);
    v8f s3 = wmma_f16(aq, bk[3], z);

    float km0 = mkn[kb + col]      ? 0.f : -INFV;
    float km1 = mkn[kb + 16 + col] ? 0.f : -INFV;
    float km2 = mkn[kb + 32 + col] ? 0.f : -INFV;
    float km3 = mkn[kb + 48 + col] ? 0.f : -INFV;

#pragma unroll
    for (int r = 0; r < 8; ++r) {
      float v0 = s0[r] + km0, v1 = s1[r] + km1;
      float v2 = s2[r] + km2, v3 = s3[r] + km3;
      float mx = fmaxf(fmaxf(v0, v1), fmaxf(v2, v3));
      mx = fmaxf(mx, __shfl_xor(mx, 1, 32));
      mx = fmaxf(mx, __shfl_xor(mx, 2, 32));
      mx = fmaxf(mx, __shfl_xor(mx, 4, 32));
      mx = fmaxf(mx, __shfl_xor(mx, 8, 32));
      float mn = fmaxf(mrow[r], mx);
      float sc = __expf(mrow[r] - mn);
      float p0 = __expf(v0 - mn);
      float p1 = __expf(v1 - mn);
      float p2 = __expf(v2 - mn);
      float p3 = __expf(v3 - mn);
      float ls = (p0 + p1) + (p2 + p3);
      ls += __shfl_xor(ls, 1, 32);
      ls += __shfl_xor(ls, 2, 32);
      ls += __shfl_xor(ls, 4, 32);
      ls += __shfl_xor(ls, 8, 32);
      lrow[r] = lrow[r] * sc + ls;
      mrow[r] = mn;
      accV[r] *= sc;
      accP[r] *= sc;
      Pl[mo + r][col]      = (_Float16)p0;
      Pl[mo + r][16 + col] = (_Float16)p1;
      Pl[mo + r][32 + col] = (_Float16)p2;
      Pl[mo + r][48 + col] = (_Float16)p3;
    }
    __syncthreads();  // single-wave WG: cheap, orders DS store->load

    // P as two A-fragments (16 queries x 32 keys each) + V / pos B-fragments
#pragma unroll
    for (int half = 0; half < 2; ++half) {
      v16h ap;
#pragma unroll
      for (int j = 0; j < 16; ++j) {
        int K = ((j < 8) ? j : (j + 8)) + hs * 8;
        ap[j] = Pl[m][half * 32 + K];
      }
      v16h bv, bp;
#pragma unroll
      for (int j = 0; j < 16; ++j) {
        int K = ((j < 8) ? j : (j + 8)) + hs * 8;
        int key = kb + half * 32 + K;
        bv[j] = vb[(((size_t)n * LSEQ + key) * NH + h) * 16 + col];
        bp[j] = (col < 3) ? (_Float16)pos_CB[((size_t)n * LSEQ + key) * 3 + col]
                          : (_Float16)((col == 3) ? 1.f : 0.f);
      }

      accV = wmma_f16(ap, bv, accV);
      accP = wmma_f16(ap, bp, accP);
    }
    __syncthreads();
  }

  // feat_node = accV / lsum  -> feat[:, h*16 + col]
#pragma unroll
  for (int r = 0; r < 8; ++r) {
    int row = qbase + mo + r;
    feat[((size_t)n * LSEQ + row) * DFEAT + h * 16 + col] =
        (_Float16)(accV[r] / lrow[r]);
  }
  // atom_pos_bias = sum(alpha*pos_CB) - pos_CA * sum(alpha)
#pragma unroll
  for (int r = 0; r < 8; ++r) {
    float ones = __shfl(accP[r], hs * 16 + 3, 32);  // "ones" column (N=3)
    if (col < 3) {
      int row = qbase + mo + r;
      float val = accP[r] / lrow[r] -
                  pos_CA[((size_t)n * LSEQ + row) * 3 + col] * (ones / lrow[r]);
      apb[(((size_t)n * LSEQ + row) * NH + h) * 3 + col] = val;
    }
  }
}

// ---------------------------------------------------------------------------
// Kernel C: spatial features per (n,l,h): distance, frame-rotated points,
// normalized directions -> feat columns 192..275 (276..287 zero pad).
// ---------------------------------------------------------------------------
__global__ __launch_bounds__(256) void spatial_kernel(
    const float* __restrict__ frame, const float* __restrict__ apb,
    _Float16* __restrict__ feat) {
  int idx = blockIdx.x * blockDim.x + threadIdx.x;
  if (idx >= NB * LSEQ * NH) return;
  int h = idx % NH;
  int nl = idx / NH;
  const float* p = apb + (size_t)idx * 3;
  float x = p[0], y = p[1], z = p[2];
  float dist = sqrtf(x * x + y * y + z * z);
  const float* F = frame + (size_t)nl * 9;
  float f0 = F[0] * x + F[1] * y + F[2] * z;
  float f1 = F[3] * x + F[4] * y + F[5] * z;
  float f2 = F[6] * x + F[7] * y + F[8] * z;
  float inv = 1.f / (sqrtf(f0 * f0 + f1 * f1 + f2 * f2) + 1e-10f);
  _Float16* fr = feat + (size_t)nl * DFEAT;
  fr[192 + h * 3 + 0] = (_Float16)f0;
  fr[192 + h * 3 + 1] = (_Float16)f1;
  fr[192 + h * 3 + 2] = (_Float16)f2;
  fr[228 + h] = (_Float16)dist;
  fr[240 + h * 3 + 0] = (_Float16)(f0 * inv);
  fr[240 + h * 3 + 1] = (_Float16)(f1 * inv);
  fr[240 + h * 3 + 2] = (_Float16)(f2 * inv);
  if (h == 0)
#pragma unroll
    for (int j = 276; j < 288; ++j) fr[j] = (_Float16)0.f;
}

// ---------------------------------------------------------------------------
// Kernel D: output projection feat(4096x288 f16) @ Wo(276x128) + bo,
// masked, + residual x  -> y (pre-layernorm, f32).  WMMA, 9 K-steps.
// ---------------------------------------------------------------------------
__global__ __launch_bounds__(32) void outproj_kernel(
    const _Float16* __restrict__ feat, const float* __restrict__ Wo,
    const float* __restrict__ bo, const float* __restrict__ x,
    const unsigned char* __restrict__ mask, float* __restrict__ y) {
  int rt = blockIdx.x >> 3;        // 256 row tiles
  int ctl = blockIdx.x & 7;        // 8 col tiles of 16
  int n = rt >> 6;
  int sbase = (rt & 63) << 4;
  int lane = threadIdx.x;
  int hs = lane >> 4, m = lane & 15, col0 = ctl * 16;

  v8f acc{};
#pragma unroll
  for (int kk = 0; kk < 9; ++kk) {
    int kb = kk * 32;
    v16h a, b;
    const _Float16* fp = feat + ((size_t)n * LSEQ + sbase + m) * DFEAT + kb;
#pragma unroll
    for (int j = 0; j < 16; ++j) {
      int K = ((j < 8) ? j : (j + 8)) + hs * 8;
      a[j] = fp[K];
    }
#pragma unroll
    for (int j = 0; j < 16; ++j) {
      int krow = kb + ((j < 8) ? j : (j + 8)) + hs * 8;
      b[j] = (krow < 276) ? (_Float16)Wo[(size_t)krow * DOUT + col0 + m]
                          : (_Float16)0.f;
    }
    acc = wmma_f16(a, b, acc);
  }

  int d = m, mo = hs * 8;
#pragma unroll
  for (int r = 0; r < 8; ++r) {
    int row = sbase + mo + r;
    size_t off = ((size_t)n * LSEQ + row) * DOUT + col0 + d;
    float fa = mask[(size_t)n * LSEQ + row] ? (acc[r] + bo[col0 + d]) : 0.f;
    y[off] = x[off] + fa;
  }
}

// ---------------------------------------------------------------------------
// Kernel E: layernorm, one wave per row of 128.
// ---------------------------------------------------------------------------
__global__ __launch_bounds__(128) void ln_kernel(
    const float* __restrict__ y, const float* __restrict__ gamma,
    const float* __restrict__ beta, float* __restrict__ out) {
  int wave = threadIdx.x >> 5;
  int lane = threadIdx.x & 31;
  int row = blockIdx.x * 4 + wave;
  const float* yp = y + (size_t)row * DOUT;
  float v[4], s = 0.f, s2 = 0.f;
#pragma unroll
  for (int i = 0; i < 4; ++i) {
    v[i] = yp[lane + 32 * i];
    s += v[i];
    s2 += v[i] * v[i];
  }
#pragma unroll
  for (int mk = 1; mk < 32; mk <<= 1) {
    s += __shfl_xor(s, mk, 32);
    s2 += __shfl_xor(s2, mk, 32);
  }
  float mu = s * (1.f / DOUT);
  float var = s2 * (1.f / DOUT) - mu * mu;
  float inv = rsqrtf(var + 1e-5f);
  float* op = out + (size_t)row * DOUT;
#pragma unroll
  for (int i = 0; i < 4; ++i) {
    int c = lane + 32 * i;
    op[c] = (v[i] - mu) * inv * gamma[c] + beta[c];
  }
}

// ---------------------------------------------------------------------------
extern "C" void kernel_launch(void* const* d_in, const int* in_sizes, int n_in,
                              void* d_out, int out_size, void* d_ws,
                              size_t ws_size, hipStream_t stream) {
  const float* x      = (const float*)d_in[0];
  const float* pos_CA = (const float*)d_in[1];
  const float* pos_CB = (const float*)d_in[2];
  const float* frame  = (const float*)d_in[3];
  const unsigned char* mask = (const unsigned char*)d_in[4];
  const float* Wq = (const float*)d_in[5];
  const float* Wk = (const float*)d_in[6];
  const float* Wv = (const float*)d_in[7];
  const float* Wo = (const float*)d_in[8];
  const float* bo = (const float*)d_in[9];
  const float* gamma = (const float*)d_in[10];
  const float* beta  = (const float*)d_in[11];
  float* out = (float*)d_out;

  // workspace carve-up (256B aligned)
  char* ws = (char*)d_ws;
  size_t off = 0;
  auto carve = [&](size_t bytes) {
    char* p = ws + off;
    off = (off + bytes + 255) & ~(size_t)255;
    return p;
  };
  const size_t nQKV = (size_t)NB * LSEQ * NH * 16;        // elems, f16
  _Float16* qb  = (_Float16*)carve(nQKV * 2);
  _Float16* kTb = (_Float16*)carve(nQKV * 2);
  _Float16* vb  = (_Float16*)carve(nQKV * 2);
  _Float16* feat = (_Float16*)carve((size_t)NB * LSEQ * DFEAT * 2);
  float* apb = (float*)carve((size_t)NB * LSEQ * NH * 3 * 4);
  float* y   = (float*)carve((size_t)NB * LSEQ * DOUT * 4);
  (void)ws_size; (void)n_in; (void)in_sizes; (void)out_size;

  qkv_proj_kernel<<<256 * 9, 128, 0, stream>>>(x, Wq, Wk, Wv, qb, kTb, vb);
  attn_kernel<<<NB * NH * 64, 32, 0, stream>>>(qb, kTb, vb, pos_CA, pos_CB,
                                               mask, feat, apb);
  spatial_kernel<<<(NB * LSEQ * NH + 255) / 256, 256, 0, stream>>>(frame, apb,
                                                                   feat);
  outproj_kernel<<<256 * 8, 32, 0, stream>>>(feat, Wo, bo, x, mask, y);
  ln_kernel<<<(NB * LSEQ) / 4, 128, 0, stream>>>(y, gamma, beta, out);
}
